// MyTransformerDecoderLayer_79671643340968
// MI455X (gfx1250) — matrix operand order repaired and checked by reading, not compiled, not run on hardware
//
#include <hip/hip_runtime.h>
#include <hip/hip_bf16.h>
#include <cstdint>
#include <cstddef>

typedef __bf16 bf16;
typedef __attribute__((ext_vector_type(16))) __bf16 v16bf;
typedef __attribute__((ext_vector_type(8)))  __bf16 v8bf;
typedef __attribute__((ext_vector_type(8)))  float  v8f;
typedef __attribute__((ext_vector_type(4)))  int    v4i;

#if defined(__AMDGCN__) && __has_builtin(__builtin_amdgcn_global_load_async_to_lds_b128) && __has_builtin(__builtin_amdgcn_s_wait_asynccnt)
#define USE_ASYNC_LDS 1
typedef __attribute__((address_space(1))) bf16* as1_bf16p;
typedef __attribute__((address_space(3))) bf16* as3_bf16p;
typedef __attribute__((address_space(1))) v4i*  as1_v4ip;
typedef __attribute__((address_space(3))) v4i*  as3_v4ip;
#else
#define USE_ASYNC_LDS 0
#endif

static __device__ __forceinline__ v8f wmma_bf16(v16bf a, v16bf b, v8f c) {
  // D = A(16x32) * B(32x16) + C, fp32 accumulate
  return __builtin_amdgcn_wmma_f32_16x16x32_bf16(false, a, false, b, (short)0, c, false, false);
}

// ---- fragment loaders (wave32 layouts per CDNA5 ISA 7.12.2) ----

// A fragment 16x32 (MxK), source row-major with leading dim `ld` (elements).
// lane m = lane&15, khalf = lane>>4; elems 0..7 -> k = khalf*8+e, 8..15 -> k = 16+khalf*8+e
static __device__ __forceinline__ v16bf load_frag_a(const bf16* p, int ld) {
  const int lane = threadIdx.x & 31;
  const bf16* r = p + (size_t)(lane & 15) * ld + (lane >> 4) * 8;
  v8bf lo = *(const v8bf*)r;
  v8bf hi = *(const v8bf*)(r + 16);
  v16bf f;
#pragma unroll
  for (int e = 0; e < 8; ++e) { f[e] = lo[e]; f[e + 8] = hi[e]; }
  return f;
}

// B fragment 32x16 (KxN) where the source holds one N-row per ld-stride row and
// K contiguous (i.e. B = src^T, src is [N, K] row-major).
// lane n = lane&15, khalf = lane>>4; elems 0..7 -> k = khalf*8+e, 8..15 -> k = 16+khalf*8+e
static __device__ __forceinline__ v16bf load_frag_b_nt(const bf16* p, int ld) {
  const int lane = threadIdx.x & 31;
  const bf16* r = p + (size_t)(lane & 15) * ld + (lane >> 4) * 8;
  v8bf lo = *(const v8bf*)r;
  v8bf hi = *(const v8bf*)(r + 16);
  v16bf f;
#pragma unroll
  for (int e = 0; e < 8; ++e) { f[e] = lo[e]; f[e + 8] = hi[e]; }
  return f;
}

// ---- elementwise f32 -> bf16, 8 elems / iteration (n must be a multiple of 8) ----
__global__ void cvt_f32_bf16(const float* __restrict__ in, bf16* __restrict__ out, size_t n8) {
  size_t i = (size_t)blockIdx.x * blockDim.x + threadIdx.x;
  size_t stride = (size_t)gridDim.x * blockDim.x;
  for (; i < n8; i += stride) {
    const float* p = in + i * 8;
    v8bf o;
#pragma unroll
    for (int e = 0; e < 8; ++e) o[e] = (bf16)p[e];
    *(v8bf*)(out + i * 8) = o;
  }
}

// ---- GEMM: C[M,N] = act((A[M,K] @ W[N,K]^T + bias) * scale) ----
// OUT_MODE: 0 -> fp32 out, 1 -> bf16 out, 2 -> bf16 out pre-transposed per head
//           (V layout: Cb[((bb*16 + h)*64 + hd) * (M/2) + t], row = t*2+bb, col = h*64+hd)
// ACT: 0 none, 1 relu.
// Block 256 = 8 waves; C tile 128x128; A/W tiles double-buffered in LDS via
// global_load_async_to_lds_b128 (ASYNCcnt ping-pong: wait <=8 keeps next tile's
// copies in flight while computing). Padded LDS rows (72) -> conflict-free
// ds_load_b128 fragment reads.
template<int OUT_MODE, int ACT>
__global__ __launch_bounds__(256) void gemm_bias(
    const bf16* __restrict__ A, const bf16* __restrict__ W,
    const float* __restrict__ bias, float scale,
    int M, int N, int K, float* __restrict__ Cf, bf16* __restrict__ Cb)
{
  constexpr int KB = 64, LDT = 72;
  __shared__ bf16 sA[2][128 * LDT];
  __shared__ bf16 sB[2][128 * LDT];
  const int tid = threadIdx.x;
  const int lane = tid & 31;
  const int wave = tid >> 5;
  const int mw = wave & 3, nw = wave >> 2;
  const int m0 = blockIdx.y * 128;
  const int n0 = blockIdx.x * 128;

  // issue the 8 per-thread copies (4 A-chunks + 4 B-chunks) for one K-tile
  auto issue_copy = [&](int buf, int k0) {
#pragma unroll
    for (int i = 0; i < 4; ++i) {
      const int idx = tid + i * 256;            // 0..1023
      const int row = idx >> 3;
      const int chk = (idx & 7) * 8;            // 8 bf16 = 16B chunks
      const bf16* ga = A + (size_t)(m0 + row) * K + k0 + chk;
      const bf16* gb = W + (size_t)(n0 + row) * K + k0 + chk;
      bf16* la = sA[buf] + row * LDT + chk;
      bf16* lb = sB[buf] + row * LDT + chk;
#if USE_ASYNC_LDS
      __builtin_amdgcn_global_load_async_to_lds_b128(
          (as1_v4ip)(as1_bf16p)const_cast<bf16*>(ga),
          (as3_v4ip)(as3_bf16p)la, 0, 0);
      __builtin_amdgcn_global_load_async_to_lds_b128(
          (as1_v4ip)(as1_bf16p)const_cast<bf16*>(gb),
          (as3_v4ip)(as3_bf16p)lb, 0, 0);
#else
      *(v8bf*)la = *(const v8bf*)ga;
      *(v8bf*)lb = *(const v8bf*)gb;
#endif
    }
  };

  v8f c[2][4];
#pragma unroll
  for (int i = 0; i < 2; ++i)
#pragma unroll
    for (int j = 0; j < 4; ++j)
#pragma unroll
      for (int r = 0; r < 8; ++r) c[i][j][r] = 0.f;

  const int nsteps = K / KB;
  issue_copy(0, 0);
  for (int step = 0; step < nsteps; ++step) {
    const int cur = step & 1;
    const bool havenext = (step + 1) < nsteps;
    // prefetch next tile into the other buffer (consumed two barriers from now)
    if (havenext) issue_copy(cur ^ 1, (step + 1) * KB);
#if USE_ASYNC_LDS
    // async ops complete in order per wave: <=8 outstanding means the current
    // tile's 8 copies have landed while the next tile's 8 stay in flight.
    if (havenext) __builtin_amdgcn_s_wait_asynccnt(8);
    else          __builtin_amdgcn_s_wait_asynccnt(0);
#endif
    __syncthreads();

    // ---- compute from LDS: two K=32 sub-steps ----
#pragma unroll
    for (int kk = 0; kk < KB; kk += 32) {
      v16bf a0 = load_frag_a(sA[cur] + (mw * 32 + 0)  * LDT + kk, LDT);
      v16bf a1 = load_frag_a(sA[cur] + (mw * 32 + 16) * LDT + kk, LDT);
#pragma unroll
      for (int j = 0; j < 4; ++j) {
        v16bf bfrag = load_frag_b_nt(sB[cur] + (nw * 64 + j * 16) * LDT + kk, LDT);
        c[0][j] = wmma_bf16(a0, bfrag, c[0][j]);
        c[1][j] = wmma_bf16(a1, bfrag, c[1][j]);
      }
    }
    __syncthreads();  // all waves done with `cur` before it is refilled
  }

  const int lg = lane >> 4, cl = lane & 15;
#pragma unroll
  for (int i = 0; i < 2; ++i) {
#pragma unroll
    for (int j = 0; j < 4; ++j) {
      const int col = n0 + nw * 64 + j * 16 + cl;
      const float bv = bias[col];
#pragma unroll
      for (int r = 0; r < 8; ++r) {
        const int row = m0 + mw * 32 + i * 16 + r + 8 * lg;
        float v = (c[i][j][r] + bv) * scale;
        if (ACT) v = fmaxf(v, 0.f);
        if (OUT_MODE == 0) {
          Cf[(size_t)row * N + col] = v;
        } else if (OUT_MODE == 1) {
          Cb[(size_t)row * N + col] = (bf16)v;
        } else {
          // per-head transpose: token t = row/2, batch bb = row&1, head h = col/64, hd = col&63
          const int t = row >> 1, bb = row & 1;
          const int h = col >> 6, hd = col & 63;
          Cb[((size_t)((bb * 16 + h) * 64 + hd)) * (size_t)(M >> 1) + t] = (bf16)v;
        }
      }
    }
  }
}

// ---- flash attention ----
// Q,K stored [len*B, D] (row = t*B + b; per (b,h) row stride = B*D = 2048).
// V is pre-transposed: Vt[(b*H+h)*HD + hd][0..Skv-1]  (row stride Skv).
// O written [T*B, D]. 128 threads = 4 waves; each wave owns 16 query rows.
// grid = (Tq/64, B*H)
#define ATT_B  2
#define ATT_H  16
#define ATT_HD 64
__global__ __launch_bounds__(128) void flash_attn(
    const bf16* __restrict__ Q, const bf16* __restrict__ K,
    const bf16* __restrict__ Vt, bf16* __restrict__ O,
    int Tq, int Skv, int causal)
{
  __shared__ bf16 pbuf[4][16 * 32];
  const int lane = threadIdx.x & 31;
  const int wave = threadIdx.x >> 5;
  const int lg = lane >> 4, cl = lane & 15;
  const int n = blockIdx.y;
  const int b = n / ATT_H, h = n % ATT_H;
  const int rs = ATT_B * ATT_H * ATT_HD;       // B*D = 2048
  const size_t hoff = (size_t)b * (ATT_H * ATT_HD) + (size_t)h * ATT_HD;
  const bf16* Qb = Q + hoff;
  const bf16* Kb = K + hoff;
  const bf16* Vh = Vt + (size_t)n * ATT_HD * (size_t)Skv;  // n == b*H+h
  bf16* Ob = O + hoff;

  const int q0 = blockIdx.x * 64 + wave * 16;

  v16bf qa0 = load_frag_a(Qb + (size_t)q0 * rs, rs);        // hd 0..31
  v16bf qa1 = load_frag_a(Qb + (size_t)q0 * rs + 32, rs);   // hd 32..63

  v16bf ones;
#pragma unroll
  for (int e = 0; e < 16; ++e) ones[e] = (bf16)1.0f;

  v8f o[4];
  float m[8], l[8];
#pragma unroll
  for (int j = 0; j < 4; ++j)
#pragma unroll
    for (int r = 0; r < 8; ++r) o[j][r] = 0.f;
#pragma unroll
  for (int r = 0; r < 8; ++r) { m[r] = -1e30f; l[r] = 0.f; }

  const int s_end = causal ? min(Skv, q0 + 16) : Skv;

  for (int s0 = 0; s0 < s_end; s0 += 32) {
    // scores: S[q, s] = sum_hd Q[q,hd] * K[s,hd]  (two 16x16 col tiles)
    v8f s[2];
#pragma unroll
    for (int j = 0; j < 2; ++j) {
#pragma unroll
      for (int r = 0; r < 8; ++r) s[j][r] = 0.f;
      v16bf kb0 = load_frag_b_nt(Kb + (size_t)(s0 + j * 16) * rs, rs);       // hd 0..31
      v16bf kb1 = load_frag_b_nt(Kb + (size_t)(s0 + j * 16) * rs + 32, rs);  // hd 32..63
      s[j] = wmma_bf16(qa0, kb0, s[j]);
      s[j] = wmma_bf16(qa1, kb1, s[j]);
    }
    if (causal) {
#pragma unroll
      for (int j = 0; j < 2; ++j) {
        const int col = s0 + j * 16 + cl;
#pragma unroll
        for (int r = 0; r < 8; ++r) {
          const int row = q0 + r + 8 * lg;
          if (col > row) s[j][r] += -1e9f;
        }
      }
    }
    // row max (16-lane butterfly within each half) + online rescale factors
    float fac[8];
#pragma unroll
    for (int r = 0; r < 8; ++r) {
      float mx = fmaxf(s[0][r], s[1][r]);
#pragma unroll
      for (int off = 8; off > 0; off >>= 1) mx = fmaxf(mx, __shfl_xor(mx, off, 32));
      const float mn = fmaxf(m[r], mx);
      fac[r] = __expf(m[r] - mn);
      m[r] = mn;
      s[0][r] = __expf(s[0][r] - mn);
      s[1][r] = __expf(s[1][r] - mn);
    }

    // relayout P (C layout fp32) -> A fragment (bf16) via per-wave LDS
    bf16* pb = pbuf[wave];
#pragma unroll
    for (int j = 0; j < 2; ++j)
#pragma unroll
      for (int r = 0; r < 8; ++r)
        pb[(r + 8 * lg) * 32 + j * 16 + cl] = (bf16)s[j][r];
    // LDS ops are in-order within a wave; compiler inserts dscnt wait on use.
    v16bf pa = load_frag_a(pb, 32);

    // row sums via WMMA: P(16x32) @ ones(32x16) -> every column = rowsum
    v8f srow;
#pragma unroll
    for (int r = 0; r < 8; ++r) srow[r] = 0.f;
    srow = wmma_bf16(pa, ones, srow);
#pragma unroll
    for (int r = 0; r < 8; ++r) l[r] = l[r] * fac[r] + srow[r];

    // rescale running output, then O += P(16x32) @ V(32x64)
#pragma unroll
    for (int jh = 0; jh < 4; ++jh)
#pragma unroll
      for (int r = 0; r < 8; ++r) o[jh][r] *= fac[r];
#pragma unroll
    for (int jh = 0; jh < 4; ++jh) {
      // Vt rows are hd (contiguous along s): two 16-byte per-lane loads, 16 elements apart
      v16bf vb = load_frag_b_nt(Vh + (size_t)(jh * 16) * Skv + s0, Skv);
      o[jh] = wmma_bf16(pa, vb, o[jh]);
    }
  }

  // normalize and store bf16
#pragma unroll
  for (int jh = 0; jh < 4; ++jh)
#pragma unroll
    for (int r = 0; r < 8; ++r) {
      const int t = q0 + r + 8 * lg;
      Ob[(size_t)t * rs + jh * 16 + cl] = (bf16)(o[jh][r] / l[r]);
    }
}

// ---- fused residual + LayerNorm over D=1024; writes fp32 and optional bf16 ----
__global__ __launch_bounds__(256) void residual_ln_1024(
    const float* __restrict__ x, const float* __restrict__ res,
    const float* __restrict__ gamma, const float* __restrict__ beta,
    float* __restrict__ yf, bf16* __restrict__ yb)
{
  const int D = 1024;
  const int row = blockIdx.x;
  const float* xr = x + (size_t)row * D;
  const float* rr = res + (size_t)row * D;
  const int c0 = threadIdx.x * 4;           // contiguous 16B chunk per thread
  float v[4];
  float sum = 0.f, ss = 0.f;
#pragma unroll
  for (int i = 0; i < 4; ++i) {
    const float t = xr[c0 + i] + rr[c0 + i];
    v[i] = t; sum += t; ss += t * t;
  }
#pragma unroll
  for (int off = 16; off > 0; off >>= 1) {
    sum += __shfl_xor(sum, off, 32);
    ss  += __shfl_xor(ss, off, 32);
  }
  __shared__ float s1[8], s2[8];
  const int wave = threadIdx.x >> 5, lane = threadIdx.x & 31;
  if (lane == 0) { s1[wave] = sum; s2[wave] = ss; }
  __syncthreads();
  sum = 0.f; ss = 0.f;
#pragma unroll
  for (int wv = 0; wv < 8; ++wv) { sum += s1[wv]; ss += s2[wv]; }
  const float mean = sum * (1.f / 1024.f);
  const float var = ss * (1.f / 1024.f) - mean * mean;
  const float inv = rsqrtf(var + 1e-5f);
#pragma unroll
  for (int i = 0; i < 4; ++i) {
    const float t = (v[i] - mean) * inv * gamma[c0 + i] + beta[c0 + i];
    yf[(size_t)row * D + c0 + i] = t;
    if (yb) yb[(size_t)row * D + c0 + i] = (bf16)t;
  }
}

// ---- orchestration ----
extern "C" void kernel_launch(void* const* d_in, const int* in_sizes, int n_in,
                              void* d_out, int out_size, void* d_ws, size_t ws_size,
                              hipStream_t stream) {
  (void)in_sizes; (void)n_in; (void)out_size; (void)ws_size;
  constexpr int T = 2048, S = 2048, B = 2, D = 1024, H = 16, FF = 4096;
  constexpr int MB = T * B;                 // 4096 rows (== S*B)
  const float scaleq = 0.125f;              // HD^-0.5 = 1/8

  const float* tgt   = (const float*)d_in[0];
  const float* memin = (const float*)d_in[1];
  // d_in[2] (tgt_mask) applied analytically (causal -1e9)
  const float* sa_qw = (const float*)d_in[3];  const float* sa_qb = (const float*)d_in[4];
  const float* sa_kw = (const float*)d_in[5];  const float* sa_kb = (const float*)d_in[6];
  const float* sa_vw = (const float*)d_in[7];  const float* sa_vb = (const float*)d_in[8];
  const float* sa_ow = (const float*)d_in[9];  const float* sa_ob = (const float*)d_in[10];
  const float* ca_qw = (const float*)d_in[11]; const float* ca_qb = (const float*)d_in[12];
  const float* ca_kw = (const float*)d_in[13]; const float* ca_kb = (const float*)d_in[14];
  const float* ca_vw = (const float*)d_in[15]; const float* ca_vb = (const float*)d_in[16];
  const float* ca_ow = (const float*)d_in[17]; const float* ca_ob = (const float*)d_in[18];
  const float* w1    = (const float*)d_in[19]; const float* b1    = (const float*)d_in[20];
  const float* w2    = (const float*)d_in[21]; const float* b2    = (const float*)d_in[22];
  const float* ln1g  = (const float*)d_in[23]; const float* ln1b  = (const float*)d_in[24];
  const float* ln2g  = (const float*)d_in[25]; const float* ln2b  = (const float*)d_in[26];
  const float* ln3g  = (const float*)d_in[27]; const float* ln3b  = (const float*)d_in[28];

  uintptr_t base = (uintptr_t)d_ws;
  auto take = [&](size_t bytes) -> void* {
    void* p = (void*)base;
    base += (bytes + 255) & ~(size_t)255;
    return p;
  };
  bf16* tgt_bf = (bf16*)take((size_t)MB * D * 2);
  bf16* mem_bf = (bf16*)take((size_t)MB * D * 2);
  bf16* wbf[8];
  for (int i = 0; i < 8; ++i) wbf[i] = (bf16*)take((size_t)D * D * 2);
  bf16* w1_bf  = (bf16*)take((size_t)FF * D * 2);
  bf16* w2_bf  = (bf16*)take((size_t)D * FF * 2);
  bf16* qbuf   = (bf16*)take((size_t)MB * D * 2);
  bf16* kbuf   = (bf16*)take((size_t)MB * D * 2);
  bf16* vbuf   = (bf16*)take((size_t)MB * D * 2);  // holds Vt: [B*H*HD, tokens]
  bf16* obuf   = (bf16*)take((size_t)MB * D * 2);
  bf16* hbuf   = (bf16*)take((size_t)MB * FF * 2);
  float* projf = (float*)take((size_t)MB * D * 4);
  float* y1f   = (float*)take((size_t)MB * D * 4);
  float* y2f   = (float*)take((size_t)MB * D * 4);
  bf16* y1b    = (bf16*)take((size_t)MB * D * 2);
  bf16* y2b    = (bf16*)take((size_t)MB * D * 2);

  auto cvt = [&](const float* in, bf16* out, size_t n) {
    size_t n8 = n / 8;                       // all sizes are multiples of 8
    int blocks = (int)((n8 + 255) / 256);
    if (blocks > 4096) blocks = 4096;
    cvt_f32_bf16<<<blocks, 256, 0, stream>>>(in, out, n8);
  };
  cvt(tgt, tgt_bf, (size_t)MB * D);
  cvt(memin, mem_bf, (size_t)MB * D);
  const float* ws_[8] = {sa_qw, sa_kw, sa_vw, sa_ow, ca_qw, ca_kw, ca_vw, ca_ow};
  for (int i = 0; i < 8; ++i) cvt(ws_[i], wbf[i], (size_t)D * D);
  cvt(w1, w1_bf, (size_t)FF * D);
  cvt(w2, w2_bf, (size_t)D * FF);

  auto gemm_bf = [&](const bf16* A, const bf16* W, const float* bias, float sc,
                     int M, int N, int K, bf16* out) {
    dim3 grid(N / 128, M / 128);
    gemm_bias<1, 0><<<grid, 256, 0, stream>>>(A, W, bias, sc, M, N, K, nullptr, out);
  };
  auto gemm_vT = [&](const bf16* A, const bf16* W, const float* bias,
                     int M, int N, int K, bf16* out) {
    dim3 grid(N / 128, M / 128);
    gemm_bias<2, 0><<<grid, 256, 0, stream>>>(A, W, bias, 1.0f, M, N, K, nullptr, out);
  };
  auto gemm_bf_relu = [&](const bf16* A, const bf16* W, const float* bias,
                          int M, int N, int K, bf16* out) {
    dim3 grid(N / 128, M / 128);
    gemm_bias<1, 1><<<grid, 256, 0, stream>>>(A, W, bias, 1.0f, M, N, K, nullptr, out);
  };
  auto gemm_f32 = [&](const bf16* A, const bf16* W, const float* bias,
                      int M, int N, int K, float* out) {
    dim3 grid(N / 128, M / 128);
    gemm_bias<0, 0><<<grid, 256, 0, stream>>>(A, W, bias, 1.0f, M, N, K, out, nullptr);
  };

  // ---- self attention (causal) ----
  gemm_bf(tgt_bf, wbf[0], sa_qb, scaleq, MB, D, D, qbuf);
  gemm_bf(tgt_bf, wbf[1], sa_kb, 1.0f,  MB, D, D, kbuf);
  gemm_vT(tgt_bf, wbf[2], sa_vb, MB, D, D, vbuf);
  flash_attn<<<dim3(T / 64, B * H), 128, 0, stream>>>(qbuf, kbuf, vbuf, obuf, T, T, 1);
  gemm_f32(obuf, wbf[3], sa_ob, MB, D, D, projf);
  residual_ln_1024<<<MB, 256, 0, stream>>>(projf, tgt, ln1g, ln1b, y1f, y1b);

  // ---- cross attention ----
  gemm_bf(y1b,    wbf[4], ca_qb, scaleq, MB, D, D, qbuf);
  gemm_bf(mem_bf, wbf[5], ca_kb, 1.0f,  MB, D, D, kbuf);
  gemm_vT(mem_bf, wbf[6], ca_vb, MB, D, D, vbuf);
  flash_attn<<<dim3(T / 64, B * H), 128, 0, stream>>>(qbuf, kbuf, vbuf, obuf, T, S, 0);
  gemm_f32(obuf, wbf[7], ca_ob, MB, D, D, projf);
  residual_ln_1024<<<MB, 256, 0, stream>>>(projf, y1f, ln2g, ln2b, y2f, y2b);

  // ---- FFN ----
  gemm_bf_relu(y2b, w1_bf, b1, MB, FF, D, hbuf);
  gemm_f32(hbuf, w2_bf, b2, MB, D, FF, projf);
  residual_ln_1024<<<MB, 256, 0, stream>>>(projf, y2f, ln3g, ln3b, (float*)d_out, nullptr);
}
